// Encoder_79645873537260
// MI455X (gfx1250) — matrix operand order repaired and checked by reading, not compiled, not run on hardware
//
#include <hip/hip_runtime.h>
#include <hip/hip_bf16.h>
#include <math.h>

typedef __attribute__((ext_vector_type(16))) _Float16 v16h;
typedef __attribute__((ext_vector_type(8)))  _Float16 v8h;
typedef __attribute__((ext_vector_type(4)))  _Float16 v4h;
typedef __attribute__((ext_vector_type(8)))  float    v8f;
typedef __attribute__((ext_vector_type(4)))  float    v4f;
typedef __attribute__((ext_vector_type(4)))  int      v4i;
typedef __attribute__((ext_vector_type(2)))  int      v2i;

#define DEV static __device__ __forceinline__

DEV v16h concat8(v8h lo, v8h hi) {
  return __builtin_shufflevector(lo, hi, 0, 1, 2, 3, 4, 5, 6, 7, 8, 9, 10, 11,
                                 12, 13, 14, 15);
}

DEV v8f wmma_f16(v16h a, v16h b, v8f c) {
  // D = A(16x32 f16) * B(32x16 f16) + C(16x16 f32)
  return __builtin_amdgcn_wmma_f32_16x16x32_f16(false, a, false, b, (short)0, c,
                                                false, false);
}

// ---- CDNA5 async global->LDS copies (ASYNCcnt), with safe fallback --------
DEV void async_cp16(const _Float16* g, _Float16* l) {
#if __has_builtin(__builtin_amdgcn_global_load_async_to_lds_b128)
  __builtin_amdgcn_global_load_async_to_lds_b128(
      (__attribute__((address_space(1))) v4i*)g,
      (__attribute__((address_space(3))) v4i*)l, 0, 0);
#else
  *(v8h*)l = *(const v8h*)g;
#endif
}

DEV void async_cp8(const _Float16* g, _Float16* l) {
#if __has_builtin(__builtin_amdgcn_global_load_async_to_lds_b64)
  __builtin_amdgcn_global_load_async_to_lds_b64(
      (__attribute__((address_space(1))) v2i*)g,
      (__attribute__((address_space(3))) v2i*)l, 0, 0);
#else
  *(v4h*)l = *(const v4h*)g;
#endif
}

DEV void async_wait() {
#if __has_builtin(__builtin_amdgcn_s_wait_asynccnt)
  __builtin_amdgcn_s_wait_asynccnt(0);
#elif __has_builtin(__builtin_amdgcn_global_load_async_to_lds_b128)
  asm volatile("s_wait_asynccnt 0" ::: "memory");
#endif
}

// A fragment 16x32 f16 from row-major [M][K] source, ld in elements.
// lanes 0-15: M=lane, K {0..7}U{16..23}; lanes 16-31: M=lane-16, K {8..15}U{24..31}.
DEV v16h load_frag_a(const _Float16* s, int ld) {
  int lane = (int)(threadIdx.x & 31u);
  int m = lane & 15;
  int kb = (lane >> 4) << 3;
  const _Float16* r = s + (size_t)m * ld + kb;
  return concat8(*(const v8h*)r, *(const v8h*)(r + 16));
}

// B fragment 32x16 f16 from K-contiguous source: s[n*ld + k] == B[k][n].
DEV v16h load_frag_bk(const _Float16* s, int ld) {
  int lane = (int)(threadIdx.x & 31u);
  int n = lane & 15;
  int kb = (lane >> 4) << 3;
  const _Float16* r = s + (size_t)n * ld + kb;
  return concat8(*(const v8h*)r, *(const v8h*)(r + 16));
}

DEV float silu_f(float x) { return x / (1.0f + __expf(-x)); }

// ---------------------------------------------------------------- utilities
// f32 [K][N] -> f16 transposed [N][Kp] (zero-padded K..Kp)
__global__ void convert_f16_t_kernel(const float* __restrict__ s,
                                     _Float16* __restrict__ d, int K, int N,
                                     int Kp) {
  int i = blockIdx.x * 256 + threadIdx.x;
  if (i < N * Kp) {
    int n = i / Kp, k = i % Kp;
    d[i] = (k < K) ? (_Float16)s[(size_t)k * N + n] : (_Float16)0.0f;
  }
}

__global__ void zero_kernel(float* __restrict__ p, int n) {
  int i = blockIdx.x * 256 + threadIdx.x;
  if (i < n) p[i] = 0.0f;
}

// pos (N,3,3) -> pos[:,1,:] (N,3)
__global__ void pos1_kernel(const float* __restrict__ pos,
                            float* __restrict__ out, int n) {
  int i = blockIdx.x * 256 + threadIdx.x;
  if (i < n * 3) {
    int a = i / 3, c = i % 3;
    out[i] = pos[a * 9 + 3 + c];
  }
}

// -------------------------------------------------------------- edge build
__global__ __launch_bounds__(256) void build_edges_kernel(
    const float* __restrict__ pos, int* __restrict__ dstIdx, unsigned salt) {
  __shared__ float sv[256];
  __shared__ int si[256];
  __shared__ float pi[3];
  int i = blockIdx.x, t = threadIdx.x;
  if (t < 3) pi[t] = pos[i * 3 + t];
  __syncthreads();
  float dist[16];
#pragma unroll
  for (int c = 0; c < 16; ++c) {
    int j = t + (c << 8);
    float dx = pi[0] - pos[j * 3];
    float dy = pi[1] - pos[j * 3 + 1];
    float dz = pi[2] - pos[j * 3 + 2];
    dist[c] = sqrtf(dx * dx + dy * dy + dz * dz);
  }
  unsigned used = 0;
  for (int e = 0; e < 20; ++e) {  // KNN (incl. self)
    float best = 3.0e38f; int bj = 0x7fffffff;
#pragma unroll
    for (int c = 0; c < 16; ++c)
      if (!((used >> c) & 1u) && dist[c] < best) { best = dist[c]; bj = t + (c << 8); }
    sv[t] = best; si[t] = bj;
    __syncthreads();
    for (int s = 128; s > 0; s >>= 1) {
      if (t < s && sv[t + s] < sv[t]) { sv[t] = sv[t + s]; si[t] = si[t + s]; }
      __syncthreads();
    }
    int win = si[0];
    if (t == 0) dstIdx[i * 60 + e] = win;
    if ((win & 255) == t) used |= 1u << (win >> 8);
    __syncthreads();
  }
  float sc[16];
#pragma unroll
  for (int c = 0; c < 16; ++c) {  // Gumbel-perturbed logits
    int j = t + (c << 8);
    unsigned x = (unsigned)i * 0x9E3779B9u ^ (unsigned)j * 0x85EBCA6Bu ^
                 salt * 0xC2B2AE35u;
    x ^= x >> 16; x *= 0x7FEB352Du; x ^= x >> 15; x *= 0x846CA68Bu; x ^= x >> 16;
    float u = (float)(x >> 8) * (1.0f / 16777216.0f) + 1e-9f;
    float gum = -__logf(-__logf(u));
    sc[c] = -3.0f * __logf(fmaxf(dist[c], 1e-10f)) + gum;
  }
  for (int e = 20; e < 60; ++e) {
    float best = -3.0e38f; int bj = 0x7fffffff;
#pragma unroll
    for (int c = 0; c < 16; ++c)
      if (!((used >> c) & 1u) && sc[c] > best) { best = sc[c]; bj = t + (c << 8); }
    sv[t] = best; si[t] = bj;
    __syncthreads();
    for (int s = 128; s > 0; s >>= 1) {
      if (t < s && sv[t + s] > sv[t]) { sv[t] = sv[t + s]; si[t] = si[t + s]; }
      __syncthreads();
    }
    int win = si[0];
    if (t == 0) dstIdx[i * 60 + e] = win;
    if ((win & 255) == t) used |= 1u << (win >> 8);
    __syncthreads();
  }
}

// ------------------------------------------------------- generic WMMA GEMM
// out = act(A @ B + bias) + resid.  A: f32 (optionally concat of A0|A1),
// Btg: f16 transposed [N][K] (K multiple of 32).  Tile 64x128, BK=32, 8 waves.
// M must be a multiple of 64 (true for all call sites).
__global__ __launch_bounds__(256) void gemm_kernel(
    const float* __restrict__ A0, const float* __restrict__ A1,
    const _Float16* __restrict__ Btg, const float* __restrict__ bias,
    const float* __restrict__ resid, float* __restrict__ outF,
    _Float16* __restrict__ outH, int M, int N, int K, int concatK, int act) {
  (void)M;
  __shared__ __align__(16) _Float16 As[64][32];
  __shared__ __align__(16) _Float16 Bt[128][40];
  int t = threadIdx.x;
  int lane = t & 31, w = t >> 5;
  int wr = w & 3, wc = w >> 2;
  int rowBase = blockIdx.y * 64;
  int colBase = blockIdx.x * 128;
  int ar = t >> 2, ak0 = (t & 3) * 8;      // A staging: row, k-run
  int bc = t & 127, bseg = (t >> 7) * 16;  // B staging: col, k-run
  v8f acc[4];
#pragma unroll
  for (int q = 0; q < 4; ++q)
#pragma unroll
    for (int e = 0; e < 8; ++e) acc[q][e] = 0.0f;

  int kTiles = K >> 5;
  for (int kt = 0; kt < kTiles; ++kt) {
    int k0 = kt << 5;
    {  // stage B: async global->LDS (2x b128 per thread)
      const _Float16* g = Btg + (size_t)(colBase + bc) * K + k0 + bseg;
      async_cp16(g, &Bt[bc][bseg]);
      async_cp16(g + 8, &Bt[bc][bseg + 8]);
      __builtin_prefetch(g + 32, 0, 1);
    }
    {  // stage A: one 8-wide f32 run -> cvt -> one v8h store
      int row = rowBase + ar, k = k0 + ak0;
      const float* src;
      if (concatK > 0)
        src = (k < concatK) ? (A0 + (size_t)row * concatK + k)
                            : (A1 + (size_t)row * (K - concatK) + (k - concatK));
      else
        src = A0 + (size_t)row * K + k;
      v4f x0 = *(const v4f*)src;
      v4f x1 = *(const v4f*)(src + 4);
      v8h hb;
#pragma unroll
      for (int j = 0; j < 4; ++j) { hb[j] = (_Float16)x0[j]; hb[4 + j] = (_Float16)x1[j]; }
      *(v8h*)(&As[ar][ak0]) = hb;
    }
    async_wait();
    __syncthreads();
    v16h af = load_frag_a(&As[wr * 16][0], 32);
#pragma unroll
    for (int nt = 0; nt < 4; ++nt) {
      v16h bf = load_frag_bk(&Bt[wc * 64 + nt * 16][0], 40);
      acc[nt] = wmma_f16(af, bf, acc[nt]);
    }
    __syncthreads();
  }
  int hl = lane >> 4, n = lane & 15;
#pragma unroll
  for (int nt = 0; nt < 4; ++nt) {
    int col = colBase + wc * 64 + nt * 16 + n;
#pragma unroll
    for (int vv = 0; vv < 8; ++vv) {
      int row = rowBase + wr * 16 + hl * 8 + vv;
      float x = acc[nt][vv];
      if (bias) x += bias[col];
      if (act == 1) x = silu_f(x);
      if (resid) x += resid[(size_t)row * N + col];
      if (outF) outF[(size_t)row * N + col] = x;
      if (outH) outH[(size_t)row * N + col] = (_Float16)x;
    }
  }
}

// --------------------------------------------------- fused EGNN edge kernel
// Per block: 64 edges. e_in = [h[src]|h[dst]|radial] (K=513, padded 544) ->
// silu GEMM1 -> silu GEMM2 -> sigmoid gate -> atomic segment-sum into agg.
// W1t: f16 [256][544] transposed+padded; W2t: f16 [256][256] transposed.
__global__ __launch_bounds__(256) void egnn_edge_kernel(
    const float* __restrict__ h, const float* __restrict__ pos,
    const int* __restrict__ dstIdx, const _Float16* __restrict__ W1t,
    const float* __restrict__ b1, const _Float16* __restrict__ W2t,
    const float* __restrict__ b2, const float* __restrict__ attW,
    const float* __restrict__ attB, float* __restrict__ agg) {
  __shared__ __align__(16) _Float16 As[64][32];
  __shared__ __align__(16) _Float16 Bt[256][40];
  __shared__ __align__(16) _Float16 Ms[64][256];
  __shared__ float gate[64];
  const int D = 256;
  int t = threadIdx.x;
  int lane = t & 31, w = t >> 5;
  int wr = w & 3, wc = w >> 2;
  int rowBase = blockIdx.x * 64;
  int ar = t >> 2, ak0 = (t & 3) * 8;
  int aeid = rowBase + ar;
  int asN = aeid / 60;
  int adN = dstIdx[aeid];

  v8f acc[8];
#pragma unroll
  for (int q = 0; q < 8; ++q)
#pragma unroll
    for (int e = 0; e < 8; ++e) acc[q][e] = 0.0f;

  for (int kt = 0; kt < 17; ++kt) {  // GEMM1, K=513 (padded 544)
    int k0 = kt << 5;
    {  // stage W1 slice: each thread one column, 32 contiguous k (async)
      const _Float16* g = W1t + (size_t)t * 544 + k0;
#pragma unroll
      for (int j = 0; j < 4; ++j) async_cp16(g + j * 8, &Bt[t][j * 8]);
      __builtin_prefetch(g + 32, 0, 1);
    }
    {  // gather-stage A (8-aligned runs never straddle the 256-split)
      int k = k0 + ak0;
      v8h hb;
      if (k < 2 * D) {
        const float* src = (k < D) ? (h + (size_t)asN * D + k)
                                   : (h + (size_t)adN * D + (k - D));
        v4f x0 = *(const v4f*)src;
        v4f x1 = *(const v4f*)(src + 4);
#pragma unroll
        for (int j = 0; j < 4; ++j) { hb[j] = (_Float16)x0[j]; hb[4 + j] = (_Float16)x1[j]; }
      } else {
#pragma unroll
        for (int j = 0; j < 8; ++j) {
          float v = 0.0f;
          if (k + j == 2 * D) {
            float dx = pos[asN * 3]     - pos[adN * 3];
            float dy = pos[asN * 3 + 1] - pos[adN * 3 + 1];
            float dz = pos[asN * 3 + 2] - pos[adN * 3 + 2];
            v = dx * dx + dy * dy + dz * dz;
          }
          hb[j] = (_Float16)v;
        }
      }
      *(v8h*)(&As[ar][ak0]) = hb;
    }
    async_wait();
    __syncthreads();
    v16h af = load_frag_a(&As[wr * 16][0], 32);
#pragma unroll
    for (int nt = 0; nt < 8; ++nt) {
      v16h bf = load_frag_bk(&Bt[wc * 128 + nt * 16][0], 40);
      acc[nt] = wmma_f16(af, bf, acc[nt]);
    }
    __syncthreads();
  }
  int hl = lane >> 4, n = lane & 15;
#pragma unroll
  for (int nt = 0; nt < 8; ++nt) {  // bias + silu -> Ms
    int col = wc * 128 + nt * 16 + n;
#pragma unroll
    for (int vv = 0; vv < 8; ++vv)
      Ms[wr * 16 + hl * 8 + vv][col] = (_Float16)silu_f(acc[nt][vv] + b1[col]);
  }
  __syncthreads();

  v8f acc2[8];
#pragma unroll
  for (int q = 0; q < 8; ++q)
#pragma unroll
    for (int e = 0; e < 8; ++e) acc2[q][e] = 0.0f;
  for (int kt = 0; kt < 8; ++kt) {  // GEMM2, K=256, A = Ms (LDS)
    int k0 = kt << 5;
    {
      const _Float16* g = W2t + (size_t)t * 256 + k0;
#pragma unroll
      for (int j = 0; j < 4; ++j) async_cp16(g + j * 8, &Bt[t][j * 8]);
      __builtin_prefetch(g + 32, 0, 1);
    }
    async_wait();
    __syncthreads();
    v16h af = load_frag_a(&Ms[wr * 16][k0], 256);
#pragma unroll
    for (int nt = 0; nt < 8; ++nt) {
      v16h bf = load_frag_bk(&Bt[wc * 128 + nt * 16][0], 40);
      acc2[nt] = wmma_f16(af, bf, acc2[nt]);
    }
    __syncthreads();
  }
  __syncthreads();
#pragma unroll
  for (int nt = 0; nt < 8; ++nt) {  // bias + silu -> Ms (reuse)
    int col = wc * 128 + nt * 16 + n;
#pragma unroll
    for (int vv = 0; vv < 8; ++vv)
      Ms[wr * 16 + hl * 8 + vv][col] = (_Float16)silu_f(acc2[nt][vv] + b2[col]);
  }
  __syncthreads();
  if (t < 64) {  // attention gate per edge row
    float dot = 0.0f;
    for (int k = 0; k < D; k += 8) {
      v8h mm = *(const v8h*)(&Ms[t][k]);
#pragma unroll
      for (int j = 0; j < 8; ++j) dot += (float)mm[j] * attW[k + j];
    }
    gate[t] = 1.0f / (1.0f + __expf(-(dot + attB[0])));
  }
  __syncthreads();
  for (int r = 0; r < 64; ++r) {  // gated segment-sum (src = eid/60)
    int eid = rowBase + r;
    int sN = eid / 60;
    atomicAdd(&agg[(size_t)sN * D + t], (float)Ms[r][t] * gate[r]);
  }
}

// ------------------------------------------------------ flash cross-MHA
// grid (8 heads, M/128). 8 waves x 16-row query tiles; K/V blocks of 32
// staged in LDS (V transposed); one WMMA per 16-key QK^T tile.
__global__ __launch_bounds__(256) void mha_kernel(
    const _Float16* __restrict__ Qsrc, const _Float16* __restrict__ KVsrc,
    float* __restrict__ out, int M, int S) {
  (void)M;
  __shared__ __align__(16) _Float16 Kb[32][32];  // [key][dim]
  __shared__ __align__(16) _Float16 Vt[32][32];  // [dim][key]
  __shared__ __align__(16) _Float16 Pb[8][16][32];
  int t = threadIdx.x;
  int lane = t & 31, w = t >> 5;
  int head = blockIdx.x;
  int qbase = blockIdx.y * 128 + w * 16;
  const float scl = 0.1767766953f;  // 1/sqrt(32)
  v16h qf = load_frag_a(Qsrc + (size_t)qbase * 768 + head * 32, 768);
  float rm[8], rl[8];
  v8f O0, O1;
#pragma unroll
  for (int v = 0; v < 8; ++v) { rm[v] = -3.0e38f; rl[v] = 0.0f; O0[v] = 0.0f; O1[v] = 0.0f; }
  int hl = lane >> 4, n = lane & 15;
  int kkey = t >> 3, kd4 = (t & 7) * 4;  // K staging
  int vdc = t & 31, vk4 = (t >> 5) * 4;  // V staging (transposed)

  for (int kb = 0; kb < S / 32; ++kb) {
    __syncthreads();
    {  // K: async b64 copy of 4 contiguous dims of one key
      const _Float16* g =
          KVsrc + (size_t)(kb * 32 + kkey) * 768 + 256 + head * 32 + kd4;
      async_cp8(g, &Kb[kkey][kd4]);
    }
    {  // V: gather 4 keys of one dim -> b64 LDS store (transposed)
      v4h vv4;
#pragma unroll
      for (int j = 0; j < 4; ++j)
        vv4[j] = KVsrc[(size_t)(kb * 32 + vk4 + j) * 768 + 512 + head * 32 + vdc];
      *(v4h*)(&Vt[vdc][vk4]) = vv4;
    }
    async_wait();
    __syncthreads();
    v8f zf;
#pragma unroll
    for (int e = 0; e < 8; ++e) zf[e] = 0.0f;
    v8f S0 = wmma_f16(qf, load_frag_bk(&Kb[0][0], 32), zf);
    v8f S1 = wmma_f16(qf, load_frag_bk(&Kb[16][0], 32), zf);
#pragma unroll
    for (int v = 0; v < 8; ++v) {  // online softmax per row (16-lane groups)
      float s0 = S0[v] * scl, s1 = S1[v] * scl;
      float mv = fmaxf(s0, s1);
#pragma unroll
      for (int off = 8; off >= 1; off >>= 1) mv = fmaxf(mv, __shfl_xor(mv, off, 32));
      float mnew = fmaxf(rm[v], mv);
      float corr = __expf(rm[v] - mnew);
      float p0 = __expf(s0 - mnew), p1 = __expf(s1 - mnew);
      float ps = p0 + p1;
#pragma unroll
      for (int off = 8; off >= 1; off >>= 1) ps += __shfl_xor(ps, off, 32);
      rl[v] = rl[v] * corr + ps;
      O0[v] *= corr; O1[v] *= corr;
      rm[v] = mnew;
      Pb[w][hl * 8 + v][n]      = (_Float16)p0;
      Pb[w][hl * 8 + v][16 + n] = (_Float16)p1;
    }
    v16h pf = load_frag_a(&Pb[w][0][0], 32);
    O0 = wmma_f16(pf, load_frag_bk(&Vt[0][0], 32), O0);
    O1 = wmma_f16(pf, load_frag_bk(&Vt[16][0], 32), O1);
  }
#pragma unroll
  for (int v = 0; v < 8; ++v) {
    float inv = 1.0f / rl[v];
    size_t row = (size_t)(qbase + hl * 8 + v);
    out[row * 256 + head * 32 + n]      = O0[v] * inv;
    out[row * 256 + head * 32 + 16 + n] = O1[v] * inv;
  }
}

// ----------------------------------------------------------- layer norm
__global__ __launch_bounds__(256) void ln_kernel(
    const float* __restrict__ x, const float* __restrict__ res,
    const float* __restrict__ g, const float* __restrict__ b,
    float* __restrict__ out) {
  __shared__ float s1[256], s2[256];
  int row = blockIdx.x, t = threadIdx.x;
  float v = x[(size_t)row * 256 + t];
  if (res) v += res[(size_t)row * 256 + t];
  s1[t] = v; s2[t] = v * v;
  __syncthreads();
  for (int s = 128; s > 0; s >>= 1) {
    if (t < s) { s1[t] += s1[t + s]; s2[t] += s2[t + s]; }
    __syncthreads();
  }
  float mu = s1[0] * (1.0f / 256.0f);
  float var = s2[0] * (1.0f / 256.0f) - mu * mu;
  out[(size_t)row * 256 + t] = (v - mu) * rsqrtf(var + 1e-5f) * g[t] + b[t];
}

// ------------------------------------------------------------------ host
extern "C" void kernel_launch(void* const* d_in, const int* in_sizes, int n_in,
                              void* d_out, int out_size, void* d_ws,
                              size_t ws_size, hipStream_t stream) {
  (void)in_sizes; (void)n_in; (void)out_size; (void)ws_size;
  const int N = 4096, D = 256, E = N * 60;
  const float* X[2]   = { (const float*)d_in[0], (const float*)d_in[1] };
  const float* POS[2] = { (const float*)d_in[2], (const float*)d_in[3] };
  auto P = [&](int idx) { return (const float*)d_in[idx]; };
  // param leaf order (jax pytree, dict keys sorted):
  // 4 + L*10: att_b, att_w, e_b1, e_b2, e_w1, e_w2, n_b1, n_b2, n_w1, n_w2
  // 44..47: ff b1,b2,w1,w2 ; 48..53: ln1_b,ln1_g,ln2_b,ln2_g,ln3_b,ln3_g
  // 54..57: mha in_b,in_w,out_b,out_w

  char* wsb = (char*)d_ws;
  size_t off = 0;
  auto alloc = [&](size_t bytes) -> void* {
    off = (off + 255) & ~(size_t)255;
    void* p = wsb + off;
    off += bytes;
    return p;
  };
  // transposed f16 weights: [N][Kp]
  _Float16 *ew1h[4], *ew2h[4], *nw1h[4], *nw2h[4];
  for (int L = 0; L < 4; ++L) {
    ew1h[L] = (_Float16*)alloc((size_t)256 * 544 * 2);
    ew2h[L] = (_Float16*)alloc((size_t)256 * 256 * 2);
    nw1h[L] = (_Float16*)alloc((size_t)256 * 512 * 2);
    nw2h[L] = (_Float16*)alloc((size_t)256 * 256 * 2);
  }
  _Float16* inwh  = (_Float16*)alloc((size_t)768 * 256 * 2);
  _Float16* outwh = (_Float16*)alloc((size_t)256 * 256 * 2);
  _Float16* ffw1h = (_Float16*)alloc((size_t)512 * 256 * 2);
  _Float16* ffw2h = (_Float16*)alloc((size_t)256 * 512 * 2);
  float* pos1[2] = { (float*)alloc((size_t)N * 3 * 4), (float*)alloc((size_t)N * 3 * 4) };
  int* dstIdx[2] = { (int*)alloc((size_t)E * 4), (int*)alloc((size_t)E * 4) };
  float* h0[2]  = { (float*)alloc((size_t)N * D * 4), (float*)alloc((size_t)N * D * 4) };
  float* h1[2]  = { (float*)alloc((size_t)N * D * 4), (float*)alloc((size_t)N * D * 4) };
  float* hn[2]  = { (float*)alloc((size_t)N * D * 4), (float*)alloc((size_t)N * D * 4) };
  float* hn2[2] = { (float*)alloc((size_t)N * D * 4), (float*)alloc((size_t)N * D * 4) };
  float* att[2] = { (float*)alloc((size_t)N * D * 4), (float*)alloc((size_t)N * D * 4) };
  _Float16* qkvh[2] = { (_Float16*)alloc((size_t)N * 768 * 2),
                        (_Float16*)alloc((size_t)N * 768 * 2) };
  float* agg   = (float*)alloc((size_t)N * D * 4);
  float* upd   = (float*)alloc((size_t)N * D * 4);
  float* tmp   = (float*)alloc((size_t)N * D * 4);
  float* ffmid = (float*)alloc((size_t)N * 512 * 4);

  auto convT = [&](const float* s, _Float16* dt, int K, int Nn, int Kp) {
    int cnt = Nn * Kp;
    convert_f16_t_kernel<<<dim3((cnt + 255) / 256), dim3(256), 0, stream>>>(
        s, dt, K, Nn, Kp);
  };
  for (int L = 0; L < 4; ++L) {
    int b = 4 + L * 10;
    convT(P(b + 4), ew1h[L], 513, 256, 544);
    convT(P(b + 5), ew2h[L], 256, 256, 256);
    convT(P(b + 8), nw1h[L], 512, 256, 512);
    convT(P(b + 9), nw2h[L], 256, 256, 256);
  }
  convT(P(55), inwh, 256, 768, 256);
  convT(P(57), outwh, 256, 256, 256);
  convT(P(46), ffw1h, 256, 512, 256);
  convT(P(47), ffw2h, 512, 256, 512);

  auto gemm = [&](const float* A0, const float* A1, int concatK,
                  const _Float16* Btg, const float* bias, const float* resid,
                  float* outF, _Float16* outH, int M, int Nn, int K, int act) {
    dim3 g(Nn / 128, (M + 63) / 64);
    gemm_kernel<<<g, dim3(256), 0, stream>>>(A0, A1, Btg, bias, resid, outF,
                                             outH, M, Nn, K, concatK, act);
  };

  for (int s = 0; s < 2; ++s) {
    pos1_kernel<<<dim3((N * 3 + 255) / 256), dim3(256), 0, stream>>>(POS[s],
                                                                     pos1[s], N);
    build_edges_kernel<<<dim3(N), dim3(256), 0, stream>>>(pos1[s], dstIdx[s],
                                                          42u + (unsigned)s);
    (void)hipMemcpyAsync(h0[s], X[s], (size_t)N * D * 4,
                         hipMemcpyDeviceToDevice, stream);
    float* hc = h0[s];
    float* hx = h1[s];
    for (int L = 0; L < 4; ++L) {
      int b = 4 + L * 10;
      zero_kernel<<<dim3(N * D / 256), dim3(256), 0, stream>>>(agg, N * D);
      egnn_edge_kernel<<<dim3(E / 64), dim3(256), 0, stream>>>(
          hc, pos1[s], dstIdx[s], ew1h[L], P(b + 2), ew2h[L], P(b + 3),
          P(b + 1), P(b + 0), agg);
      gemm(hc, agg, 256, nw1h[L], P(b + 6), nullptr, upd, nullptr, N, 256, 512, 1);
      gemm(upd, nullptr, 0, nw2h[L], P(b + 7), hc, hx, nullptr, N, 256, 256, 0);
      float* tsw = hc; hc = hx; hx = tsw;
    }
    ln_kernel<<<dim3(N), dim3(256), 0, stream>>>(hc, X[s], P(49), P(48), hn[s]);
    gemm(hn[s], nullptr, 0, inwh, P(54), nullptr, nullptr, qkvh[s], N, 768, 256, 0);
  }

  mha_kernel<<<dim3(8, N / 128), dim3(256), 0, stream>>>(qkvh[0], qkvh[1],
                                                         att[0], N, N);
  mha_kernel<<<dim3(8, N / 128), dim3(256), 0, stream>>>(qkvh[1], qkvh[0],
                                                         att[1], N, N);

  for (int s = 0; s < 2; ++s) {
    gemm(att[s], nullptr, 0, outwh, P(56), nullptr, tmp, nullptr, N, 256, 256, 0);
    ln_kernel<<<dim3(N), dim3(256), 0, stream>>>(tmp, hn[s], P(51), P(50), hn2[s]);
    gemm(hn2[s], nullptr, 0, ffw1h, P(44), nullptr, ffmid, nullptr, N, 512, 256, 1);
    gemm(ffmid, nullptr, 0, ffw2h, P(45), nullptr, tmp, nullptr, N, 256, 512, 0);
    ln_kernel<<<dim3(N), dim3(256), 0, stream>>>(
        tmp, hn2[s], P(53), P(52), (float*)d_out + (size_t)s * N * D);
  }
}
